// FullSelfAttention_16441134809843
// MI455X (gfx1250) — matrix operand order repaired and checked by reading, not compiled, not run on hardware
//
#include <hip/hip_runtime.h>
#include <hip/hip_bf16.h>
#include <math.h>

// ---------------- problem constants ----------------
#define BB     2
#define NN     1024
#define DIMC   1024
#define NHEADS 16
#define DHD    64
#define MEMN   4
#define SEQL   1024
#define JJ     (MEMN*SEQL + NN)   // 5120
#define INNER  (NHEADS*DHD)       // 1024
#define JDIFF  (JJ - NN)          // 4096

typedef __bf16 bf16_t;
typedef __attribute__((ext_vector_type(16))) __bf16 v16bf;
typedef __attribute__((ext_vector_type(8)))  float  v8f;

// =====================================================================
// Prep kernels
// =====================================================================
__global__ void prep_x(const float* __restrict__ x, bf16_t* __restrict__ xb,
                       float* __restrict__ newmem) {
  size_t i = (size_t)blockIdx.x * blockDim.x + threadIdx.x;
  if (i >= (size_t)BB * NN * DIMC) return;
  float v = x[i];
  xb[i] = (bf16_t)v;
  newmem[i] = v;               // new_mem = x[:, -seq_len:] == x here
}

__global__ void prep_kx(const float* __restrict__ memi, const float* __restrict__ x,
                        bf16_t* __restrict__ kxb) {
  size_t i = (size_t)blockIdx.x * blockDim.x + threadIdx.x;
  if (i >= (size_t)BB * JJ * DIMC) return;
  size_t b = i / ((size_t)JJ * DIMC);
  size_t rem = i % ((size_t)JJ * DIMC);
  size_t j = rem / DIMC, d = rem % DIMC;
  float v = (j < (size_t)(MEMN*SEQL))
              ? memi[(b * (size_t)(MEMN*SEQL) + j) * DIMC + d]
              : x[(b * (size_t)NN + (j - MEMN*SEQL)) * DIMC + d];
  kxb[i] = (bf16_t)v;
}

// Wt[n*DIMC + k] = W[k*INNER + n]  (column-major bf16 copy of weight)
__global__ void prep_wT(const float* __restrict__ W, bf16_t* __restrict__ Wt) {
  int i = blockIdx.x * blockDim.x + threadIdx.x;
  if (i >= DIMC * INNER) return;
  int n = i >> 10, k = i & 1023;
  Wt[i] = (bf16_t)W[(size_t)k * INNER + n];
}

// bias[h*JJ + j] = rel_table[rel_idxs[sel]*NHEADS + h]
__global__ void prep_bias(const int* __restrict__ rel_idxs,
                          const float* __restrict__ rel_table,
                          float* __restrict__ bias) {
  int i = blockIdx.x * blockDim.x + threadIdx.x;
  if (i >= NHEADS * JJ) return;
  int h = i / JJ, j = i % JJ;
  int ridx = (j < MEMN*SEQL) ? rel_idxs[j >> 10] : rel_idxs[MEMN];
  bias[i] = rel_table[(size_t)ridx * NHEADS + h];
}

// =====================================================================
// bf16 WMMA GEMM, 32x32 output per wave (2x2 tiles of 16x16):
//   C[M x 1024] = A[M x 1024] @ Bt^T     (Bt column-major: Bt[n*1024+k])
// Per k-step of 32: 4 fragment loads feed 4 WMMAs (2x reuse of A and B).
// mode 0: bf16 out (Q)       mode 1: bf16 out + rel bias (K)
// mode 2: bf16 out, transposed vT[b,h,d,j] (V)
// mode 3: fp32 out + bo (final projection)
// =====================================================================
__global__ __launch_bounds__(128) void gemm32_bf16(
    const bf16_t* __restrict__ A, const bf16_t* __restrict__ Bt,
    void* __restrict__ Out, int M, int mode,
    const float* __restrict__ bias, const float* __restrict__ bo)
{
  const int lane = threadIdx.x & 31;
  const int wv   = threadIdx.x >> 5;
  const int l16  = lane & 15;
  const int g    = lane >> 4;                        // half of the wave
  const int nt32 = ((blockIdx.x & 7) << 2) | wv;     // 0..31 (32-col supertile)
  const int mt32 = blockIdx.x >> 3;
  const int row0 = mt32 * 32, col0 = nt32 * 32;

  const bf16_t* Ar0 = A  + (size_t)(row0 + l16) * DIMC;
  const bf16_t* Ar1 = Ar0 + (size_t)16 * DIMC;
  const bf16_t* Br0 = Bt + (size_t)(col0 + l16) * DIMC;
  const bf16_t* Br1 = Br0 + (size_t)16 * DIMC;

  v8f acc[2][2] = {{{}, {}}, {{}, {}}};
  #pragma unroll 2
  for (int kk = 0; kk < DIMC; kk += 32) {
    union F { v16bf v; uint4 q[2]; } ua0, ua1, ub0, ub1;
    // A-frag (16x32): lane m=l16; e<8 -> k=g*8+e ; e>=8 -> k=16+g*8+(e-8)
    ua0.q[0] = *(const uint4*)(Ar0 + kk + g*8);
    ua0.q[1] = *(const uint4*)(Ar0 + kk + 16 + g*8);
    ua1.q[0] = *(const uint4*)(Ar1 + kk + g*8);
    ua1.q[1] = *(const uint4*)(Ar1 + kk + 16 + g*8);
    // B-frag (32x16): lane n=l16; e<8 -> k=g*8+e ; e>=8 -> k=16+g*8+(e-8)
    ub0.q[0] = *(const uint4*)(Br0 + kk + g*8);
    ub0.q[1] = *(const uint4*)(Br0 + kk + 16 + g*8);
    ub1.q[0] = *(const uint4*)(Br1 + kk + g*8);
    ub1.q[1] = *(const uint4*)(Br1 + kk + 16 + g*8);
    acc[0][0] = __builtin_amdgcn_wmma_f32_16x16x32_bf16(false, ua0.v, false, ub0.v,
                                                        (short)0, acc[0][0], false, false);
    acc[0][1] = __builtin_amdgcn_wmma_f32_16x16x32_bf16(false, ua0.v, false, ub1.v,
                                                        (short)0, acc[0][1], false, false);
    acc[1][0] = __builtin_amdgcn_wmma_f32_16x16x32_bf16(false, ua1.v, false, ub0.v,
                                                        (short)0, acc[1][0], false, false);
    acc[1][1] = __builtin_amdgcn_wmma_f32_16x16x32_bf16(false, ua1.v, false, ub1.v,
                                                        (short)0, acc[1][1], false, false);
  }

  #pragma unroll
  for (int sm = 0; sm < 2; ++sm) {
    #pragma unroll
    for (int sn = 0; sn < 2; ++sn) {
      const int col = col0 + sn*16 + l16;
      #pragma unroll
      for (int r = 0; r < 8; ++r) {
        int mrow = row0 + sm*16 + r + 8*g;   // C layout: M = r + 8*(lane>>4)
        float v = acc[sm][sn][r];
        if (mode == 0) {
          ((bf16_t*)Out)[(size_t)mrow * INNER + col] = (bf16_t)v;
        } else if (mode == 1) {
          int j = mrow % JJ;
          int h = col >> 6;
          v += bias[h * JJ + j];
          ((bf16_t*)Out)[(size_t)mrow * INNER + col] = (bf16_t)v;
        } else if (mode == 2) {
          int b = mrow / JJ, j = mrow % JJ;
          int h = col >> 6, d = col & 63;
          ((bf16_t*)Out)[(((size_t)(b*NHEADS + h))*DHD + d) * JJ + j] = (bf16_t)v;
        } else {
          ((float*)Out)[(size_t)mrow * DIMC + col] = v + bo[col];
        }
      }
    }
  }
}

// =====================================================================
// Flash attention, one wave per (b, h, 32-row i-supertile) = two 16-i tiles
// sharing every K / V fragment load.
//   S^T = K_tile(16j x 64d) @ Q^T(64d x 16i)  -> softmax rows in lanes
//   O^T += V^T(16d x 32j) @ P^T(32j x 16i)    (4 d-tiles)
// =====================================================================
__global__ __launch_bounds__(128) void attn_fwd_wmma(
    const bf16_t* __restrict__ qb, const bf16_t* __restrict__ kb,
    const bf16_t* __restrict__ vT, const unsigned char* __restrict__ maskp,
    bf16_t* __restrict__ ao)
{
  const int lane = threadIdx.x & 31;
  const int l16 = lane & 15, g = lane >> 4;
  const int w   = blockIdx.x * 4 + (threadIdx.x >> 5);
  const int it32s = NN / 32;               // 32
  const int b   = w / (NHEADS * it32s);
  const int rem = w % (NHEADS * it32s);
  const int h   = rem / it32s;
  const int i0  = (rem % it32s) * 32;

  // Q^T B-fragments: u = i-subtile (0/1), kk = d-chunk (0..31 / 32..63)
  union F { v16bf v; uint4 q[2]; };
  F bq[2][2];
  #pragma unroll
  for (int u = 0; u < 2; ++u) {
    const bf16_t* qrow = qb + ((size_t)(b*NN + i0 + u*16 + l16)) * INNER + h*DHD;
    #pragma unroll
    for (int kk = 0; kk < 2; ++kk) {
      bq[u][kk].q[0] = *(const uint4*)(qrow + kk*32 + g*8);
      bq[u][kk].q[1] = *(const uint4*)(qrow + kk*32 + 16 + g*8);
    }
  }

  v8f acco[2][4] = {{{}, {}, {}, {}}, {{}, {}, {}, {}}};
  float mrun[2] = {-3.0e38f, -3.0e38f};
  float lrun[2] = {0.0f, 0.0f};
  int   ig[2];
  ig[0] = i0 + l16;                        // global query row (within N)
  ig[1] = i0 + 16 + l16;

  int jend = i0 + 31 + JDIFF + 1;          // causal upper bound (exclusive)
  if (jend > JJ) jend = JJ;
  jend = (jend + 31) & ~31;

  const unsigned char* mb = maskp + (size_t)b * JJ;
  const bf16_t* vbase = vT + ((size_t)(b*NHEADS + h)) * DHD * JJ;

  for (int jt = 0; jt < jend; jt += 32) {
    // ---- S^T for two 16-j sub-tiles, both i-subtiles share K fragments ----
    v8f sac[2][2] = {{{}, {}}, {{}, {}}};
    #pragma unroll
    for (int t = 0; t < 2; ++t) {
      const bf16_t* krow = kb + ((size_t)(b*JJ + jt + t*16 + l16)) * INNER + h*DHD;
      #pragma unroll
      for (int kk = 0; kk < 2; ++kk) {
        F ua;
        ua.q[0] = *(const uint4*)(krow + kk*32 + g*8);
        ua.q[1] = *(const uint4*)(krow + kk*32 + 16 + g*8);
        #pragma unroll
        for (int u = 0; u < 2; ++u)
          sac[u][t] = __builtin_amdgcn_wmma_f32_16x16x32_bf16(false, ua.v, false, bq[u][kk].v,
                                                              (short)0, sac[u][t], false, false);
      }
    }

    // ---- mask bytes shared across the two i-subtiles ----
    float mval[16];
    #pragma unroll
    for (int t = 0; t < 2; ++t)
      #pragma unroll
      for (int r = 0; r < 8; ++r)
        mval[t*8 + r] = mb[jt + t*16 + 8*g + r] ? 1.0f : 0.0f;

    v16bf pf[2];
    #pragma unroll
    for (int u = 0; u < 2; ++u) {
      // s[t*8+r] holds S^T[jt + t*16 + 8*g + r, i = l16] for subtile u
      float s[16];
      #pragma unroll
      for (int t = 0; t < 2; ++t)
        #pragma unroll
        for (int r = 0; r < 8; ++r) {
          int jg = jt + t*16 + 8*g + r;
          float v = sac[u][t][r] * 0.125f;       // dh^-0.5
          bool dead = (jg > ig[u] + JDIFF) || (mval[t*8 + r] != 0.0f);
          s[t*8 + r] = dead ? -3.0e38f : v;
        }

      // online softmax per query row (i = l16)
      float lm = s[0];
      #pragma unroll
      for (int e = 1; e < 16; ++e) lm = fmaxf(lm, s[e]);
      lm = fmaxf(lm, __shfl_xor(lm, 16, 32));
      float mnew = fmaxf(mrun[u], lm);
      float corr = __expf(mrun[u] - mnew);
      float p[16], ls = 0.0f;
      #pragma unroll
      for (int e = 0; e < 16; ++e) { p[e] = __expf(s[e] - mnew); ls += p[e]; }
      ls += __shfl_xor(ls, 16, 32);
      lrun[u] = lrun[u] * corr + ls;
      mrun[u] = mnew;
      #pragma unroll
      for (int t = 0; t < 4; ++t)
        #pragma unroll
        for (int r = 0; r < 8; ++r) acco[u][t][r] *= corr;

      // rebuild P^T as B-fragment (cross-half exchange):
      // lane(g) keeps tile-g regs (e = g*8+r), receives other half's tile-g
      // regs (e = (1-g)*8+r).
      #pragma unroll
      for (int r = 0; r < 8; ++r) {
        float keep = g ? p[8 + r] : p[r];
        float send = g ? p[r]     : p[8 + r];
        float recv = __shfl_xor(send, 16, 32);
        pf[u][r]     = (bf16_t)(g ? recv : keep);
        pf[u][8 + r] = (bf16_t)(g ? keep : recv);
      }
    }

    // ---- O^T += V^T @ P^T (4 d-tiles); V fragments shared across u ----
    #pragma unroll
    for (int t = 0; t < 4; ++t) {
      const bf16_t* vrow = vbase + (size_t)(t*16 + l16) * JJ + jt;
      F uv;
      uv.q[0] = *(const uint4*)(vrow + g*16);
      uv.q[1] = *(const uint4*)(vrow + g*16 + 8);
      #pragma unroll
      for (int u = 0; u < 2; ++u)
        acco[u][t] = __builtin_amdgcn_wmma_f32_16x16x32_bf16(false, uv.v, false, pf[u],
                                                             (short)0, acco[u][t], false, false);
    }
  }

  // ---- finalize: divide by row sum, write bf16 attn output (row-major) ----
  #pragma unroll
  for (int u = 0; u < 2; ++u) {
    float inv = 1.0f / lrun[u];
    bf16_t* aorow = ao + ((size_t)(b*NN + i0 + u*16 + l16)) * INNER + h*DHD;
    #pragma unroll
    for (int t = 0; t < 4; ++t)
      #pragma unroll
      for (int r = 0; r < 8; ++r)
        aorow[t*16 + 8*g + r] = (bf16_t)(acco[u][t][r] * inv);
  }
}

// =====================================================================
// Host launcher
// =====================================================================
extern "C" void kernel_launch(void* const* d_in, const int* in_sizes, int n_in,
                              void* d_out, int out_size, void* d_ws, size_t ws_size,
                              hipStream_t stream) {
  (void)in_sizes; (void)n_in; (void)out_size; (void)ws_size;

  const float* x    = (const float*)d_in[0];
  const float* memi = (const float*)d_in[1];
  const unsigned char* mask = (const unsigned char*)d_in[2];
  const int*   rel_idxs  = (const int*)d_in[3];
  const float* Wq = (const float*)d_in[4];
  const float* Wk = (const float*)d_in[5];
  const float* Wv = (const float*)d_in[6];
  const float* Wo = (const float*)d_in[7];
  const float* bo = (const float*)d_in[8];
  const float* rel_table = (const float*)d_in[9];

  float* outp   = (float*)d_out;
  float* newmem = outp + (size_t)BB * NN * DIMC;

  // workspace carve-up (~84 MB total)
  char* ws = (char*)d_ws;
  size_t off = 0;
  auto carve = [&](size_t bytes) {
    char* p = ws + off;
    off += (bytes + 255) & ~(size_t)255;
    return p;
  };
  bf16_t* xb   = (bf16_t*)carve((size_t)BB*NN*DIMC * 2);
  bf16_t* kxb  = (bf16_t*)carve((size_t)BB*JJ*DIMC * 2);
  bf16_t* WqT  = (bf16_t*)carve((size_t)DIMC*INNER * 2);
  bf16_t* WkT  = (bf16_t*)carve((size_t)DIMC*INNER * 2);
  bf16_t* WvT  = (bf16_t*)carve((size_t)DIMC*INNER * 2);
  bf16_t* WoT  = (bf16_t*)carve((size_t)DIMC*INNER * 2);
  bf16_t* qb   = (bf16_t*)carve((size_t)BB*NN*INNER * 2);
  bf16_t* kb   = (bf16_t*)carve((size_t)BB*JJ*INNER * 2);
  bf16_t* vTb  = (bf16_t*)carve((size_t)BB*JJ*INNER * 2);
  bf16_t* aob  = (bf16_t*)carve((size_t)BB*NN*INNER * 2);
  float*  bias = (float*) carve((size_t)NHEADS*JJ * 4);

  // ---- prep ----
  {
    int n = BB*NN*DIMC;
    prep_x<<<(n + 255)/256, 256, 0, stream>>>(x, xb, newmem);
  }
  {
    size_t n = (size_t)BB*JJ*DIMC;
    prep_kx<<<(unsigned)((n + 255)/256), 256, 0, stream>>>(memi, x, kxb);
  }
  {
    int n = DIMC*INNER;
    prep_wT<<<(n + 255)/256, 256, 0, stream>>>(Wq, WqT);
    prep_wT<<<(n + 255)/256, 256, 0, stream>>>(Wk, WkT);
    prep_wT<<<(n + 255)/256, 256, 0, stream>>>(Wv, WvT);
    prep_wT<<<(n + 255)/256, 256, 0, stream>>>(Wo, WoT);
  }
  {
    int n = NHEADS*JJ;
    prep_bias<<<(n + 255)/256, 256, 0, stream>>>(rel_idxs, rel_table, bias);
  }

  // ---- projections (bf16 WMMA GEMMs, 32x32 per wave) ----
  {
    int M = BB*NN;   // 2048 -> 64 m-supertiles, 8 blocks per m-row
    gemm32_bf16<<<(M/32)*8, 128, 0, stream>>>(xb, WqT, qb, M, 0, nullptr, nullptr);
  }
  {
    int M = BB*JJ;   // 10240 -> 320 m-supertiles
    gemm32_bf16<<<(M/32)*8, 128, 0, stream>>>(kxb, WkT, kb,  M, 1, bias, nullptr);
    gemm32_bf16<<<(M/32)*8, 128, 0, stream>>>(kxb, WvT, vTb, M, 2, nullptr, nullptr);
  }

  // ---- flash attention (32 query rows per wave) ----
  {
    int waves = BB * NHEADS * (NN/32);   // 1024
    attn_fwd_wmma<<<waves/4, 128, 0, stream>>>(qb, kb, vTb, mask, aob);
  }

  // ---- output projection (fp32 out + bo) ----
  {
    int M = BB*NN;
    gemm32_bf16<<<(M/32)*8, 128, 0, stream>>>(aob, WoT, outp, M, 3, nullptr, bo);
  }
}
